// RowSelfAttention_71614284693902
// MI455X (gfx1250) — matrix operand order repaired and checked by reading, not compiled, not run on hardware
//
#include <hip/hip_runtime.h>

typedef __attribute__((ext_vector_type(8)))  __bf16 v8bf;
typedef __attribute__((ext_vector_type(16))) __bf16 v16bf;
typedef __attribute__((ext_vector_type(8)))  float  v8f;
typedef __attribute__((ext_vector_type(4))) unsigned int u32x4;
typedef __attribute__((ext_vector_type(4))) int          i32x4;
typedef __attribute__((ext_vector_type(8))) int          i32x8;

#define BATCH 2
#define EMB   768
#define NH    12
#define HD    64
#define ROWS  128
#define COLS  256
#define NPOS  (ROWS * COLS)   // 32768 spatial positions
#define KDIM  (HD * ROWS)     // 8192  tied-attention K dimension
#define MQKV  (3 * EMB)       // 2304  fused QKV output channels

#define WMMA_BF16(acc, a, b) \
    acc = __builtin_amdgcn_wmma_f32_16x16x32_bf16(false, (a), false, (b), (short)0, (acc), false, false)

// ---------- helpers ----------

__device__ __forceinline__ __bf16 f2bf(float f) {
    unsigned u = __builtin_bit_cast(unsigned, f);
    u += 0x7FFFu + ((u >> 16) & 1u);          // round-to-nearest-even
    unsigned short s = (unsigned short)(u >> 16);
    return __builtin_bit_cast(__bf16, s);
}

// A-fragment: 16x32 bf16 tile (MxK), row-major source, ld in elements.
__device__ __forceinline__ v16bf load_a(const __bf16* __restrict__ tile, int ld, int lane) {
    const int half = (lane >> 4) & 1;
    const int m    = lane & 15;
    const __bf16* p = tile + (size_t)m * (size_t)ld + half * 8;
    v8bf lo = *reinterpret_cast<const v8bf*>(p);
    v8bf hi = *reinterpret_cast<const v8bf*>(p + 16);
    return __builtin_shufflevector(lo, hi, 0,1,2,3,4,5,6,7,8,9,10,11,12,13,14,15);
}

// B-fragment: 32x16 bf16 (KxN), memory holds B^T row-major ([N][K], contiguous K).
__device__ __forceinline__ v16bf load_b(const __bf16* __restrict__ tile, int ld, int lane) {
    const int half = (lane >> 4) & 1;
    const int n    = lane & 15;
    const __bf16* p = tile + (size_t)n * (size_t)ld + half * 8;
    v8bf lo = *reinterpret_cast<const v8bf*>(p);
    v8bf hi = *reinterpret_cast<const v8bf*>(p + 16);
    return __builtin_shufflevector(lo, hi, 0,1,2,3,4,5,6,7,8,9,10,11,12,13,14,15);
}

// Tensor Data Mover: 2-D tile load (rows x 32 K-elements, bf16) global -> LDS.
// D# bit layout per CDNA5 ISA ch.8: group0 {count, lds_addr, global_addr, type=2},
// group1 {data_size=2B, tensor_dim0=ldk, tensor_dim1=rows, tile_dim0=32,
//         tile_dim1=rows, tensor_dim0_stride=ldk}.
__device__ __forceinline__ void tdm_load_2d(__bf16* lds_dst, const __bf16* gsrc,
                                            int rows, int ldk) {
    const unsigned long long ga = (unsigned long long)(uintptr_t)gsrc;
    const unsigned lds = (unsigned)(uintptr_t)lds_dst;   // LDS aperture: low 32 bits = offset
    const unsigned d0 = (unsigned)ldk;
    const unsigned d1 = (unsigned)rows;
    u32x4 g0 = { 1u,                                          // count=1, user descriptor
                 lds,                                         // lds_addr
                 (unsigned)(ga & 0xffffffffull),              // global_addr[31:0]
                 (unsigned)((ga >> 32) & 0x01ffffffull) | (2u << 30) };  // [56:32] | type=2
    i32x8 g1 = { (int)(1u << 16),                                          // data_size=1 (2B)
                 (int)((d0 & 0xffffu) << 16),                              // tensor_dim0 lo
                 (int)(((d0 >> 16) & 0xffffu) | ((d1 & 0xffffu) << 16)),   // dim0 hi | dim1 lo
                 (int)(((d1 >> 16) & 0xffffu) | (32u << 16)),              // dim1 hi | tile_dim0=32
                 (int)(d1 & 0xffffu),                                      // tile_dim1=rows
                 (int)d0,                                                  // tensor_dim0_stride lo
                 0, 0 };
    i32x4 gz = { 0, 0, 0, 0 };
#if defined(__clang_major__) && (__clang_major__ >= 23)
    i32x8 gz8 = { 0, 0, 0, 0, 0, 0, 0, 0 };
    __builtin_amdgcn_tensor_load_to_lds(g0, g1, gz, gz, gz8, 0);
#else
    __builtin_amdgcn_tensor_load_to_lds(g0, g1, gz, gz, 0);
#endif
}

// Shared TDM-staged GEMM mainloop: block tile 64(M) x 128(N), double-buffered
// LDS panels filled by the Tensor Data Mover (wave 0), 8 waves each computing a
// 32x32 register tile (2x2 WMMAs) from ds_load_b128 fragments.
__device__ __forceinline__ void gemm_stage_loop(const __bf16* __restrict__ Abase, int ldka,
                                                const __bf16* __restrict__ Bbase, int ldkb,
                                                __bf16* As /* [2][64*32] */,
                                                __bf16* Bs /* [2][128*32] */,
                                                int nk, int wave, int lane,
                                                v8f (&acc)[2][2]) {
    if (wave == 0) {
        tdm_load_2d(As, Abase, 64, ldka);
        tdm_load_2d(Bs, Bbase, 128, ldkb);
        __builtin_amdgcn_s_wait_tensorcnt(0);
    }
    __syncthreads();
    const int mw = (wave & 1) * 32;
    const int nw = (wave >> 1) * 32;
    int cur = 0;
    for (int ki = 0; ki < nk; ++ki) {
        const int nxt = cur ^ 1;
        if (ki + 1 < nk && wave == 0) {
            tdm_load_2d(As + nxt * (64 * 32), Abase + (size_t)(ki + 1) * 32, 64, ldka);
            tdm_load_2d(Bs + nxt * (128 * 32), Bbase + (size_t)(ki + 1) * 32, 128, ldkb);
        }
        const __bf16* Ap = As + cur * (64 * 32);
        const __bf16* Bp = Bs + cur * (128 * 32);
        v16bf a0 = load_a(Ap + (mw +  0) * 32, 32, lane);
        v16bf a1 = load_a(Ap + (mw + 16) * 32, 32, lane);
        v16bf b0 = load_b(Bp + (nw +  0) * 32, 32, lane);
        v16bf b1 = load_b(Bp + (nw + 16) * 32, 32, lane);
        WMMA_BF16(acc[0][0], a0, b0);
        WMMA_BF16(acc[0][1], a0, b1);
        WMMA_BF16(acc[1][0], a1, b0);
        WMMA_BF16(acc[1][1], a1, b1);
        if (ki + 1 < nk && wave == 0) __builtin_amdgcn_s_wait_tensorcnt(0);
        __syncthreads();
        cur = nxt;
    }
}

// ---------- kernel 0: per-batch scaling = hd^-0.5 * eff^-0.5 ----------

__global__ __launch_bounds__(256) void k_scale(const float* __restrict__ mask,
                                               float* __restrict__ scale_ws) {
    __shared__ float cnt[BATCH];
    const int t = threadIdx.x;
    if (t < BATCH) cnt[t] = 0.0f;
    __syncthreads();
    const int b = t >> 7;
    const int r = t & 127;
    const float* mrow = mask + ((size_t)b * ROWS + r) * COLS;
    float unp = 0.0f;
    for (int c = 0; c < COLS; ++c) unp += (1.0f - mrow[c]);
    atomicAdd(&cnt[b], (unp > 0.0f) ? 1.0f : 0.0f);
    __syncthreads();
    if (t < BATCH) scale_ws[t] = rsqrtf((float)HD) * rsqrtf(cnt[t]);
}

// ---------- kernel 1: convert four weight matrices to bf16 ----------

__global__ __launch_bounds__(256) void k_cvt_w(const float* __restrict__ w0,
                                               const float* __restrict__ w1,
                                               const float* __restrict__ w2,
                                               const float* __restrict__ w3,
                                               __bf16* __restrict__ out) {
    const int idx   = blockIdx.x * 256 + threadIdx.x;
    const int which = idx / (EMB * EMB);
    const int off   = idx % (EMB * EMB);
    const float* s  = (which == 0) ? w0 : (which == 1) ? w1 : (which == 2) ? w2 : w3;
    out[idx] = f2bf(s[off]);
}

// ---------- kernel 2: coalesced LDS tile transpose src -> xT[b][n][e] bf16 ----------
// 64x64 tiles: fp32 reads coalesced along n, bf16 writes coalesced along e.

__global__ __launch_bounds__(256) void k_xT(const float* __restrict__ src,
                                            __bf16* __restrict__ xT) {
    __shared__ __bf16 tile[64][72];                      // padded vs bank conflicts
    const int nt = blockIdx.x % (NPOS / 64);
    const int et = (blockIdx.x / (NPOS / 64)) % (EMB / 64);
    const int b  = blockIdx.x / ((NPOS / 64) * (EMB / 64));
    const int e0 = et * 64, n0 = nt * 64;
    const int tr = threadIdx.x >> 6;                     // 0..3
    const int tc = threadIdx.x & 63;
#pragma unroll
    for (int i = 0; i < 64; i += 4) {
        const int e = i + tr;
        tile[e][tc] = f2bf(src[((size_t)b * EMB + e0 + e) * NPOS + n0 + tc]);
    }
    __syncthreads();
#pragma unroll
    for (int i = 0; i < 64; i += 4) {
        const int n = i + tr;
        xT[((size_t)b * NPOS + n0 + n) * EMB + e0 + tc] = tile[tc][n];
    }
}

// ---------- kernel 3: fused QKV projection [2304x768]x[768x32768], TDM-staged ----------

__global__ __launch_bounds__(256) void k_qkv(const __bf16* __restrict__ wbf,
                                             const float* __restrict__ bq,
                                             const float* __restrict__ bk,
                                             const float* __restrict__ bv,
                                             const float* __restrict__ mask,
                                             const __bf16* __restrict__ xT,
                                             __bf16* __restrict__ qw,
                                             __bf16* __restrict__ kw,
                                             __bf16* __restrict__ vw) {
    __shared__ __bf16 As[2][64 * 32];
    __shared__ __bf16 Bs[2][128 * 32];
    const int lane = threadIdx.x & 31;
    const int wave = threadIdx.x >> 5;
    const int MT = MQKV / 64;                 // 36
    const int bm = blockIdx.x % MT;
    const int bn = (blockIdx.x / MT) % (NPOS / 128);
    const int b  = blockIdx.x / (MT * (NPOS / 128));

    const __bf16* Abase = wbf + (size_t)(bm * 64) * EMB;
    const __bf16* Bbase = xT + ((size_t)b * NPOS + bn * 128) * EMB;

    v8f acc[2][2] = {};
    gemm_stage_loop(Abase, EMB, Bbase, EMB, &As[0][0], &Bs[0][0], EMB / 32, wave, lane, acc);

    const int half  = lane >> 4;
    const int mw = (wave & 1) * 32;
    const int nw = (wave >> 1) * 32;
    const int which = (bm * 64) / EMB;        // block-uniform: q(0), k(1), v(2)
    const int obase = bm * 64 - which * EMB + mw;
    const int nbase = bn * 128 + nw;

    float qm[2]; int rr[2], cc[2];
#pragma unroll
    for (int j = 0; j < 2; ++j) {
        const int n = nbase + j * 16 + (lane & 15);
        rr[j] = n >> 8;
        cc[j] = n & 255;
        qm[j] = 1.0f - mask[((size_t)b * ROWS + rr[j]) * COLS + cc[j]];
    }

    if (which == 0) {
#pragma unroll
        for (int i = 0; i < 2; ++i)
#pragma unroll
            for (int j = 0; j < 2; ++j)
#pragma unroll
                for (int vv = 0; vv < 8; ++vv) {
                    const int o = obase + i * 16 + vv + 8 * half;
                    const int h = o >> 6, d = o & 63;
                    const size_t ix = ((size_t)(b * NH + h) * COLS + cc[j]) * KDIM + (size_t)d * ROWS + rr[j];
                    qw[ix] = f2bf((acc[i][j][vv] + bq[o]) * qm[j]);
                }
    } else if (which == 1) {
#pragma unroll
        for (int i = 0; i < 2; ++i)
#pragma unroll
            for (int j = 0; j < 2; ++j)
#pragma unroll
                for (int vv = 0; vv < 8; ++vv) {
                    const int o = obase + i * 16 + vv + 8 * half;
                    const int h = o >> 6, d = o & 63;
                    const size_t ix = ((size_t)(b * NH + h) * COLS + cc[j]) * KDIM + (size_t)d * ROWS + rr[j];
                    kw[ix] = f2bf(acc[i][j][vv] + bk[o]);
                }
    } else {
#pragma unroll
        for (int i = 0; i < 2; ++i)
#pragma unroll
            for (int j = 0; j < 2; ++j)
#pragma unroll
                for (int vv = 0; vv < 8; ++vv) {
                    const int o = obase + i * 16 + vv + 8 * half;
                    const int h = o >> 6, d = o & 63;
                    const size_t ix = ((size_t)(b * NH + h) * KDIM + (size_t)d * ROWS + rr[j]) * COLS + cc[j];
                    vw[ix] = f2bf(acc[i][j][vv] + bv[o]);
                }
    }
}

// ---------- kernel 4: tied-row scores, per (b,h) [256x256xK=8192], TDM-staged ----------

__global__ __launch_bounds__(256) void k_scores(const __bf16* __restrict__ qw,
                                                const __bf16* __restrict__ kw,
                                                const float* __restrict__ scale_ws,
                                                float* __restrict__ aw) {
    __shared__ __bf16 As[2][64 * 32];
    __shared__ __bf16 Bs[2][128 * 32];
    const int lane = threadIdx.x & 31;
    const int wave = threadIdx.x >> 5;
    const int bn = blockIdx.x % (COLS / 128);            // 2
    const int bm = (blockIdx.x / (COLS / 128)) % (COLS / 64);  // 4
    const int bh = blockIdx.x / ((COLS / 128) * (COLS / 64));

    const __bf16* Abase = qw + ((size_t)bh * COLS + bm * 64) * KDIM;
    const __bf16* Bbase = kw + ((size_t)bh * COLS + bn * 128) * KDIM;

    v8f acc[2][2] = {};
    gemm_stage_loop(Abase, KDIM, Bbase, KDIM, &As[0][0], &Bs[0][0], KDIM / 32, wave, lane, acc);

    const float s  = scale_ws[bh / NH];
    const int half = lane >> 4;
    const int mw = (wave & 1) * 32;
    const int nw = (wave >> 1) * 32;
#pragma unroll
    for (int i = 0; i < 2; ++i)
#pragma unroll
        for (int j = 0; j < 2; ++j)
#pragma unroll
            for (int vv = 0; vv < 8; ++vv) {
                const int c = bm * 64 + mw + i * 16 + vv + 8 * half;
                const int e = bn * 128 + nw + j * 16 + (lane & 15);
                aw[((size_t)bh * COLS + c) * COLS + e] = acc[i][j][vv] * s;
            }
}

// ---------- kernel 5: wave32 softmax over 256-wide rows ----------

__global__ __launch_bounds__(256) void k_softmax(const float* __restrict__ aw,
                                                 const float* __restrict__ mask,
                                                 __bf16* __restrict__ p) {
    const int lane = threadIdx.x & 31;
    const int row  = blockIdx.x * 8 + (threadIdx.x >> 5);
    const int bh   = row >> 8;
    const int b    = bh / NH;
    const float* rp = aw + (size_t)row * COLS;
    const float* cm = mask + (size_t)b * ROWS * COLS;

    float vals[8];
    float m = -INFINITY;
#pragma unroll
    for (int j = 0; j < 8; ++j) {
        const int e = lane + 32 * j;
        float x = rp[e];
        if (cm[e] != 0.0f) x = -INFINITY;
        vals[j] = x;
        m = fmaxf(m, x);
    }
    for (int off = 16; off > 0; off >>= 1) m = fmaxf(m, __shfl_xor(m, off, 32));
    float sum = 0.0f;
#pragma unroll
    for (int j = 0; j < 8; ++j) { const float e = __expf(vals[j] - m); vals[j] = e; sum += e; }
    for (int off = 16; off > 0; off >>= 1) sum += __shfl_xor(sum, off, 32);
    const float inv = 1.0f / sum;
    __bf16* op = p + (size_t)row * COLS;
#pragma unroll
    for (int j = 0; j < 8; ++j) op[lane + 32 * j] = f2bf(vals[j] * inv);
}

// ---------- kernel 6: apply probs, per (b,h) [256x8192xK=256], TDM-staged ----------

__global__ __launch_bounds__(256) void k_av(const __bf16* __restrict__ p,
                                            const __bf16* __restrict__ vw,
                                            __bf16* __restrict__ ctx) {
    __shared__ __bf16 As[2][64 * 32];
    __shared__ __bf16 Bs[2][128 * 32];
    const int lane = threadIdx.x & 31;
    const int wave = threadIdx.x >> 5;
    const int bn = blockIdx.x % (KDIM / 128);            // 64
    const int bm = (blockIdx.x / (KDIM / 128)) % (COLS / 64);  // 4
    const int bh = blockIdx.x / ((KDIM / 128) * (COLS / 64));

    const __bf16* Abase = p  + ((size_t)bh * COLS + bm * 64) * COLS;
    const __bf16* Bbase = vw + ((size_t)bh * KDIM + bn * 128) * COLS;

    v8f acc[2][2] = {};
    gemm_stage_loop(Abase, COLS, Bbase, COLS, &As[0][0], &Bs[0][0], COLS / 32, wave, lane, acc);

    const int b_ = bh / NH, h = bh % NH;
    const int half = lane >> 4;
    const int mw = (wave & 1) * 32;
    const int nw = (wave >> 1) * 32;
#pragma unroll
    for (int i = 0; i < 2; ++i)
#pragma unroll
        for (int j = 0; j < 2; ++j) {
            const int dr = bn * 128 + nw + j * 16 + (lane & 15);
            const int d  = dr >> 7, r = dr & 127;
            const int ii = h * HD + d;
#pragma unroll
            for (int vv = 0; vv < 8; ++vv) {
                const int c = bm * 64 + mw + i * 16 + vv + 8 * half;
                ctx[((size_t)b_ * NPOS + ((size_t)r << 8) + c) * EMB + ii] = f2bf(acc[i][j][vv]);
            }
        }
}

// ---------- kernel 7: output projection [768x768]x[768x32768], TDM-staged ----------

__global__ __launch_bounds__(256) void k_out(const __bf16* __restrict__ wbf,
                                             const float* __restrict__ bo,
                                             const __bf16* __restrict__ ctx,
                                             float* __restrict__ out) {
    __shared__ __bf16 As[2][64 * 32];
    __shared__ __bf16 Bs[2][128 * 32];
    const int lane = threadIdx.x & 31;
    const int wave = threadIdx.x >> 5;
    const int MT = EMB / 64;                  // 12
    const int bm = blockIdx.x % MT;
    const int bn = (blockIdx.x / MT) % (NPOS / 128);
    const int b  = blockIdx.x / (MT * (NPOS / 128));

    const __bf16* Abase = wbf + (size_t)3 * EMB * EMB + (size_t)(bm * 64) * EMB;
    const __bf16* Bbase = ctx + ((size_t)b * NPOS + bn * 128) * EMB;

    v8f acc[2][2] = {};
    gemm_stage_loop(Abase, EMB, Bbase, EMB, &As[0][0], &Bs[0][0], EMB / 32, wave, lane, acc);

    const int half = lane >> 4;
    const int mw = (wave & 1) * 32;
    const int nw = (wave >> 1) * 32;
#pragma unroll
    for (int i = 0; i < 2; ++i)
#pragma unroll
        for (int j = 0; j < 2; ++j) {
            const int n = bn * 128 + nw + j * 16 + (lane & 15);
#pragma unroll
            for (int vv = 0; vv < 8; ++vv) {
                const int o = bm * 64 + mw + i * 16 + vv + 8 * half;
                out[((size_t)b * EMB + o) * NPOS + n] = acc[i][j][vv] + bo[o];
            }
        }
}

// ---------- host launch ----------

extern "C" void kernel_launch(void* const* d_in, const int* in_sizes, int n_in,
                              void* d_out, int out_size, void* d_ws, size_t ws_size,
                              hipStream_t stream) {
    (void)in_sizes; (void)n_in; (void)out_size; (void)ws_size;
    const float* src  = (const float*)d_in[0];
    const float* mask = (const float*)d_in[1];
    const float* wq   = (const float*)d_in[2];
    const float* bq   = (const float*)d_in[3];
    const float* wk   = (const float*)d_in[4];
    const float* bk   = (const float*)d_in[5];
    const float* wv   = (const float*)d_in[6];
    const float* bv   = (const float*)d_in[7];
    const float* wo   = (const float*)d_in[8];
    const float* bo   = (const float*)d_in[9];
    float* out = (float*)d_out;

    char* base = (char*)d_ws;
    size_t off = 0;
    auto alloc = [&](size_t bytes) -> void* {
        void* p = base + off;
        off = (off + bytes + 255) & ~(size_t)255;
        return p;
    };
    float*  scale_ws = (float*) alloc((size_t)BATCH * sizeof(float));
    __bf16* wbf      = (__bf16*)alloc((size_t)4 * EMB * EMB * 2);
    __bf16* xT       = (__bf16*)alloc((size_t)BATCH * NPOS * EMB * 2);          // reused as ctx
    __bf16* qwp      = (__bf16*)alloc((size_t)BATCH * NH * COLS * KDIM * 2);
    __bf16* kwp      = (__bf16*)alloc((size_t)BATCH * NH * COLS * KDIM * 2);
    __bf16* vwp      = (__bf16*)alloc((size_t)BATCH * NH * KDIM * COLS * 2);
    float*  aw       = (float*) alloc((size_t)BATCH * NH * COLS * COLS * 4);
    __bf16* pws      = (__bf16*)alloc((size_t)BATCH * NH * COLS * COLS * 2);
    __bf16* ctx      = xT;   // xT dead after k_qkv; reuse for attention output

    k_scale  <<<1, 256, 0, stream>>>(mask, scale_ws);
    k_cvt_w  <<<(4 * EMB * EMB) / 256, 256, 0, stream>>>(wq, wk, wv, wo, wbf);
    k_xT     <<<BATCH * (EMB / 64) * (NPOS / 64), 256, 0, stream>>>(src, xT);
    k_qkv    <<<BATCH * (MQKV / 64) * (NPOS / 128), 256, 0, stream>>>(wbf, bq, bk, bv, mask, xT, qwp, kwp, vwp);
    k_scores <<<BATCH * NH * (COLS / 64) * (COLS / 128), 256, 0, stream>>>(qwp, kwp, scale_ws, aw);
    k_softmax<<<(BATCH * NH * COLS) / 8, 256, 0, stream>>>(aw, mask, pws);
    k_av     <<<BATCH * NH * (COLS / 64) * (KDIM / 128), 256, 0, stream>>>(pws, vwp, ctx);
    k_out    <<<BATCH * (EMB / 64) * (NPOS / 128), 256, 0, stream>>>(wbf, bo, ctx, out);
}
